// LocalTransformer_52467320488068
// MI455X (gfx1250) — matrix operand order repaired and checked by reading, not compiled, not run on hardware
//
#include <hip/hip_runtime.h>
#include <cstdint>
#include <cstddef>

typedef __attribute__((ext_vector_type(16))) __bf16 v16bf;
typedef __attribute__((ext_vector_type(8)))  float  v8f;

union FragU { v16bf v; uint4 q[2]; };

// ---- problem sizes (match reference) ----
constexpr int kB    = 16;
constexpr int kN    = 4096;
constexpr int kLD   = 256;
constexpr int kH    = 8;
constexpr int kDH   = 64;
constexpr int kW    = 128;
constexpr int kINNER= kH * kDH;      // 512
constexpr int kFF   = 4 * kLD;       // 1024
constexpr int kROWS = kB * kN;       // 65536
constexpr int kNW   = kN / kW;       // 32
constexpr int kSLK  = 50;
constexpr int kOUTN = kN - 2 * kSLK; // 3996

// =====================================================================
// fp32 -> bf16 conversion + transpose:  in[l][K][Nn] -> out[l][Nn][K]
// =====================================================================
__global__ void transpose_f2bf_kernel(const float* __restrict__ in,
                                      __bf16* __restrict__ out,
                                      int K, int Nn, int layers) {
    size_t i = (size_t)blockIdx.x * 256 + threadIdx.x;
    size_t per = (size_t)K * Nn;
    if (i >= per * layers) return;
    size_t l = i / per, rem = i - l * per;
    size_t k = rem / Nn, n = rem - k * Nn;
    out[l * per + n * K + k] = (__bf16)in[i];
}

// =====================================================================
// h = x @ W_exp + b_exp + pos_emb   (D_IN = 2, trivial)
// =====================================================================
__global__ void __launch_bounds__(256) expand_kernel(
    const float* __restrict__ x, const float* __restrict__ We,
    const float* __restrict__ be, const float* __restrict__ pe,
    float* __restrict__ h) {
    size_t row = blockIdx.x;
    int c = threadIdx.x;
    int n = (int)(row & (kN - 1));
    float x0 = x[row * 2 + 0], x1 = x[row * 2 + 1];
    h[row * kLD + c] = x0 * We[c] + x1 * We[kLD + c] + be[c] + pe[(size_t)n * kLD + c];
}

// =====================================================================
// LayerNorm (one wave32 per 256-wide row) -> bf16
// =====================================================================
__global__ void __launch_bounds__(256) ln_kernel(
    const float* __restrict__ x, const float* __restrict__ g,
    const float* __restrict__ bt, __bf16* __restrict__ out) {
    int wave = threadIdx.x >> 5, lane = threadIdx.x & 31;
    size_t row = (size_t)blockIdx.x * 8 + wave;
    const float* xr = x + row * kLD;
    float v[8], s = 0.f, ss = 0.f;
#pragma unroll
    for (int i = 0; i < 8; ++i) { float t = xr[lane + i * 32]; v[i] = t; s += t; ss += t * t; }
#pragma unroll
    for (int m = 1; m < 32; m <<= 1) { s += __shfl_xor(s, m, 32); ss += __shfl_xor(ss, m, 32); }
    float mean = s * (1.f / kLD);
    float var  = ss * (1.f / kLD) - mean * mean;
    float rs   = rsqrtf(var + 1e-5f);
    __bf16* orow = out + row * kLD;
#pragma unroll
    for (int i = 0; i < 8; ++i) {
        int c = lane + i * 32;
        orow[c] = (__bf16)((v[i] - mean) * rs * g[c] + bt[c]);
    }
}

// =====================================================================
// bf16 WMMA GEMM: C[M,Nn] = A[M,K] @ W[K,Nn], weights TRANSPOSED as
// WtT[Nn][K].  Block 256 thr = 8 waves -> 128(M) x 128(N) tile, each
// wave computes a 16x128 strip (8 wmma / 32-wide K-step).
// Staging: CDNA5 async global->LDS (ASYNCcnt) with double-buffered LDS
// so the next tile's fetch overlaps the current tile's WMMAs.
//   mode 0: out bf16 | mode 1: out bf16, bias+GELU | mode 2: resid+=acc(+bias)
// =====================================================================
__global__ void __launch_bounds__(256) gemm_bf16_wmma(
    const __bf16* __restrict__ A, const __bf16* __restrict__ WtT,
    const float* __restrict__ bias, float* __restrict__ resid,
    __bf16* __restrict__ outb, int M, int Nn, int K, int mode) {
    __shared__ alignas(16) __bf16 As[2][128][32];
    __shared__ alignas(16) __bf16 Bs[2][128][32];   // Bs[buf][n][k]

    const int tid  = threadIdx.x;
    const int lane = tid & 31, wave = tid >> 5;
    const int hl = lane >> 4, l16 = lane & 15;
    const int m0 = blockIdx.y * 128, n0 = blockIdx.x * 128;

    // staging coords: 256 thr x 32 bytes = one 128x32 bf16 tile
    const int sr = tid >> 1, sc = (tid & 1) * 16;
    const __bf16* gA = A   + (size_t)(m0 + sr) * K + sc;
    const __bf16* gB = WtT + (size_t)(n0 + sr) * K + sc;

    // async global->LDS: 4 x b128 per thread, tracked by ASYNCcnt
    auto issue_async = [&](const __bf16* a0, const __bf16* b0, int bi) {
        uint32_t la0 = (uint32_t)(uintptr_t)&As[bi][sr][sc];
        uint32_t la1 = (uint32_t)(uintptr_t)&As[bi][sr][sc + 8];
        uint32_t lb0 = (uint32_t)(uintptr_t)&Bs[bi][sr][sc];
        uint32_t lb1 = (uint32_t)(uintptr_t)&Bs[bi][sr][sc + 8];
        asm volatile(
            "global_load_async_to_lds_b128 %0, %4, off\n\t"
            "global_load_async_to_lds_b128 %1, %5, off\n\t"
            "global_load_async_to_lds_b128 %2, %6, off\n\t"
            "global_load_async_to_lds_b128 %3, %7, off"
            :: "v"(la0), "v"(la1), "v"(lb0), "v"(lb1),
               "v"(a0), "v"(a0 + 8), "v"(b0), "v"(b0 + 8)
            : "memory");
    };

    issue_async(gA, gB, 0);
    asm volatile("s_wait_asynccnt 0x0" ::: "memory");
    __syncthreads();

    v8f acc[8] = {{}, {}, {}, {}, {}, {}, {}, {}};

    for (int k0 = 0; k0 < K; k0 += 32) {
        const int bi = (k0 >> 5) & 1;
        if (k0 + 32 < K) {
            issue_async(gA + k0 + 32, gB + k0 + 32, bi ^ 1);   // overlap with WMMAs
            __builtin_prefetch(gA + k0 + 64, 0, 3);            // global_prefetch_b8
            __builtin_prefetch(gB + k0 + 64, 0, 3);
        }

        FragU ua;
        const int m = wave * 16 + l16;
        ua.q[0] = *(const uint4*)&As[bi][m][hl * 8];
        ua.q[1] = *(const uint4*)&As[bi][m][16 + hl * 8];
#pragma unroll
        for (int nt = 0; nt < 8; ++nt) {
            FragU ub;
            const int cn = nt * 16 + l16;
            ub.q[0] = *(const uint4*)&Bs[bi][cn][hl * 8];
            ub.q[1] = *(const uint4*)&Bs[bi][cn][16 + hl * 8];
            acc[nt] = __builtin_amdgcn_wmma_f32_16x16x32_bf16(
                false, ua.v, false, ub.v, (short)0, acc[nt], false, false);
        }
        asm volatile("s_wait_asynccnt 0x0" ::: "memory");
        __syncthreads();
    }

#pragma unroll
    for (int nt = 0; nt < 8; ++nt) {
        const int col = n0 + nt * 16 + l16;
        const float bv = bias ? bias[col] : 0.f;
#pragma unroll
        for (int r = 0; r < 8; ++r) {
            int row = m0 + wave * 16 + hl * 8 + r;   // C layout: slot r -> row r+8*half
            float v = acc[nt][r] + bv;
            size_t idx = (size_t)row * Nn + col;
            if (mode == 0) {
                outb[idx] = (__bf16)v;
            } else if (mode == 1) {
                outb[idx] = (__bf16)(0.5f * v * (1.f + erff(v * 0.70710678118654752f)));
            } else {
                resid[idx] += v;
            }
        }
    }
}

// =====================================================================
// Fused local attention: rotary + QK^T + mask + online softmax + PV
// One block (256 thr, 8 waves) per (window, head, batch).
// =====================================================================
__global__ void __launch_bounds__(256) attn_kernel(
    const __bf16* __restrict__ qkv, __bf16* __restrict__ ao) {
    const int win = blockIdx.x, hd = blockIdx.y, bb = blockIdx.z;
    __shared__ alignas(16) __bf16 Qs[128][64];
    __shared__ alignas(16) __bf16 Ks[128][64];
    __shared__ alignas(16) __bf16 VsT[64][128];   // transposed: [d][key]
    __shared__ alignas(16) __bf16 Ps[8][16][32];

    const int tid = threadIdx.x, lane = tid & 31, wave = tid >> 5;
    const int hl = lane >> 4, l16 = lane & 15;

    // ---- stage Q with rotary + 1/sqrt(DH) scale ----
#pragma unroll 4
    for (int i = 0; i < 16; ++i) {
        int idx = i * 256 + tid;                    // 128 rows x 32 pairs
        int qr = idx >> 5, dp = idx & 31;
        int pos = win * kW + qr;
        size_t base = (size_t)(bb * kN + pos) * (3 * kINNER) + hd * kDH;
        float q0 = (float)qkv[base + dp];
        float q1 = (float)qkv[base + dp + 32];
        float inv = powf(10000.f, -(float)dp * (1.f / 32.f));
        float ang = (float)pos * inv;
        float cs = cosf(ang), sn = sinf(ang);
        Qs[qr][dp]      = (__bf16)((q0 * cs - q1 * sn) * 0.125f);
        Qs[qr][dp + 32] = (__bf16)((q1 * cs + q0 * sn) * 0.125f);
    }
    __syncthreads();

    // ---- Q A-fragments (d = 0..31 and 32..63) ----
    FragU aq0, aq1;
    {
        const int m = wave * 16 + l16;
        aq0.q[0] = *(const uint4*)&Qs[m][hl * 8];
        aq0.q[1] = *(const uint4*)&Qs[m][16 + hl * 8];
        aq1.q[0] = *(const uint4*)&Qs[m][32 + hl * 8];
        aq1.q[1] = *(const uint4*)&Qs[m][48 + hl * 8];
    }

    v8f ot[4] = {{}, {}, {}, {}};
    float mrow[8], lrow[8];
#pragma unroll
    for (int r = 0; r < 8; ++r) { mrow[r] = -1e30f; lrow[r] = 0.f; }

    for (int kb = 0; kb < 2; ++kb) {            // previous window, current window
        __syncthreads();
        const int kbase = (win - 1 + kb) * kW;  // negative for win==0,kb==0
        // stage K (rotary, zero-fill OOB)
#pragma unroll 4
        for (int i = 0; i < 16; ++i) {
            int idx = i * 256 + tid;
            int kr = idx >> 5, dp = idx & 31;
            int pos = kbase + kr;
            float k0 = 0.f, k1 = 0.f;
            if (pos >= 0) {
                size_t base = (size_t)(bb * kN + pos) * (3 * kINNER) + kINNER + hd * kDH;
                k0 = (float)qkv[base + dp];
                k1 = (float)qkv[base + dp + 32];
            }
            float inv = powf(10000.f, -(float)dp * (1.f / 32.f));
            float ang = (float)(pos < 0 ? 0 : pos) * inv;
            float cs = cosf(ang), sn = sinf(ang);
            Ks[kr][dp]      = (__bf16)(k0 * cs - k1 * sn);
            Ks[kr][dp + 32] = (__bf16)(k1 * cs + k0 * sn);
        }
        // stage V transposed (zero-fill OOB)
#pragma unroll 4
        for (int i = 0; i < 32; ++i) {
            int idx = i * 256 + tid;
            int vr = idx >> 6, dc = idx & 63;
            int pos = kbase + vr;
            float vv = 0.f;
            if (pos >= 0)
                vv = (float)qkv[(size_t)(bb * kN + pos) * (3 * kINNER) + 2 * kINNER + hd * kDH + dc];
            VsT[dc][vr] = (__bf16)vv;
        }
        __syncthreads();

        for (int c = 0; c < 4; ++c) {           // 32-key chunks
            const int j0 = kbase + c * 32;
            v8f s0 = {}, s1 = {};
            FragU bk;
            // S tile 0: keys c*32 + l16
            bk.q[0] = *(const uint4*)&Ks[c * 32 + l16][hl * 8];
            bk.q[1] = *(const uint4*)&Ks[c * 32 + l16][16 + hl * 8];
            s0 = __builtin_amdgcn_wmma_f32_16x16x32_bf16(false, aq0.v, false, bk.v, (short)0, s0, false, false);
            bk.q[0] = *(const uint4*)&Ks[c * 32 + l16][32 + hl * 8];
            bk.q[1] = *(const uint4*)&Ks[c * 32 + l16][48 + hl * 8];
            s0 = __builtin_amdgcn_wmma_f32_16x16x32_bf16(false, aq1.v, false, bk.v, (short)0, s0, false, false);
            // S tile 1: keys c*32 + 16 + l16
            bk.q[0] = *(const uint4*)&Ks[c * 32 + 16 + l16][hl * 8];
            bk.q[1] = *(const uint4*)&Ks[c * 32 + 16 + l16][16 + hl * 8];
            s1 = __builtin_amdgcn_wmma_f32_16x16x32_bf16(false, aq0.v, false, bk.v, (short)0, s1, false, false);
            bk.q[0] = *(const uint4*)&Ks[c * 32 + 16 + l16][32 + hl * 8];
            bk.q[1] = *(const uint4*)&Ks[c * 32 + 16 + l16][48 + hl * 8];
            s1 = __builtin_amdgcn_wmma_f32_16x16x32_bf16(false, aq1.v, false, bk.v, (short)0, s1, false, false);

            // mask + online-softmax stats (row r+8*hl lives in acc slot r)
            const int ibase = win * kW + wave * 16 + hl * 8;
            const int jA = j0 + l16, jB = j0 + 16 + l16;
            float corr[8];
#pragma unroll
            for (int r = 0; r < 8; ++r) {
                int ii = ibase + r;
                float sa = s0[r], sb = s1[r];
                bool ma = (jA < 0) || (ii < jA) || (ii - jA > kW);
                bool mb = (jB < 0) || (ii < jB) || (ii - jB > kW);
                sa = ma ? -1e9f : sa;
                sb = mb ? -1e9f : sb;
                s0[r] = sa; s1[r] = sb;
                float t = fmaxf(sa, sb);
                t = fmaxf(t, __shfl_xor(t, 1, 32));
                t = fmaxf(t, __shfl_xor(t, 2, 32));
                t = fmaxf(t, __shfl_xor(t, 4, 32));
                t = fmaxf(t, __shfl_xor(t, 8, 32));
                float mn = fmaxf(mrow[r], t);
                corr[r] = expf(mrow[r] - mn);
                mrow[r] = mn;
            }
#pragma unroll
            for (int r = 0; r < 8; ++r) {
                float pa = expf(s0[r] - mrow[r]);
                float pb = expf(s1[r] - mrow[r]);
                float t = pa + pb;
                t += __shfl_xor(t, 1, 32);
                t += __shfl_xor(t, 2, 32);
                t += __shfl_xor(t, 4, 32);
                t += __shfl_xor(t, 8, 32);
                lrow[r] = lrow[r] * corr[r] + t;
                Ps[wave][hl * 8 + r][l16]      = (__bf16)pa;   // transpose via LDS
                Ps[wave][hl * 8 + r][16 + l16] = (__bf16)pb;
            }
#pragma unroll
            for (int t2 = 0; t2 < 4; ++t2)
#pragma unroll
                for (int r = 0; r < 8; ++r) ot[t2][r] *= corr[r];

            // P as A-fragment (16 queries x 32 keys)
            FragU pf;
            pf.q[0] = *(const uint4*)&Ps[wave][l16][hl * 8];
            pf.q[1] = *(const uint4*)&Ps[wave][l16][16 + hl * 8];
            // O += P @ V  (4 d-tiles of 16) -- B fragments contiguous from VsT
#pragma unroll
            for (int t2 = 0; t2 < 4; ++t2) {
                FragU bv;
                bv.q[0] = *(const uint4*)&VsT[t2 * 16 + l16][c * 32 + hl * 8];
                bv.q[1] = *(const uint4*)&VsT[t2 * 16 + l16][c * 32 + 16 + hl * 8];
                ot[t2] = __builtin_amdgcn_wmma_f32_16x16x32_bf16(
                    false, pf.v, false, bv.v, (short)0, ot[t2], false, false);
            }
        }
    }

    // ---- write O / l  (layout [b, n, h*dh]) ----
#pragma unroll
    for (int t2 = 0; t2 < 4; ++t2) {
#pragma unroll
        for (int r = 0; r < 8; ++r) {
            int n = win * kW + wave * 16 + hl * 8 + r;
            int dh = t2 * 16 + l16;
            size_t idx = (size_t)(bb * kN + n) * kINNER + hd * kDH + dh;
            ao[idx] = (__bf16)(ot[t2][r] / lrow[r]);
        }
    }
}

// =====================================================================
// Final LN + logits (256 -> 1) + slack crop, one wave per row
// =====================================================================
__global__ void __launch_bounds__(256) logits_kernel(
    const float* __restrict__ h, const float* __restrict__ g,
    const float* __restrict__ bt, const float* __restrict__ Wl,
    float* __restrict__ out) {
    int wave = threadIdx.x >> 5, lane = threadIdx.x & 31;
    size_t row = (size_t)blockIdx.x * 8 + wave;
    const float* xr = h + row * kLD;
    float v[8], s = 0.f, ss = 0.f;
#pragma unroll
    for (int i = 0; i < 8; ++i) { float t = xr[lane + i * 32]; v[i] = t; s += t; ss += t * t; }
#pragma unroll
    for (int m = 1; m < 32; m <<= 1) { s += __shfl_xor(s, m, 32); ss += __shfl_xor(ss, m, 32); }
    float mean = s * (1.f / kLD);
    float var  = ss * (1.f / kLD) - mean * mean;
    float rs   = rsqrtf(var + 1e-5f);
    float dot = 0.f;
#pragma unroll
    for (int i = 0; i < 8; ++i) {
        int c = lane + i * 32;
        dot += ((v[i] - mean) * rs * g[c] + bt[c]) * Wl[c];
    }
#pragma unroll
    for (int m = 1; m < 32; m <<= 1) dot += __shfl_xor(dot, m, 32);
    if (lane == 0) {
        int n = (int)(row & (kN - 1));
        if (n >= kSLK && n < kN - kSLK) {
            size_t b = row >> 12;
            out[b * kOUTN + (n - kSLK)] = dot;
        }
    }
}

// =====================================================================
extern "C" void kernel_launch(void* const* d_in, const int* in_sizes, int n_in,
                              void* d_out, int out_size, void* d_ws, size_t ws_size,
                              hipStream_t stream) {
    (void)in_sizes; (void)n_in; (void)out_size; (void)ws_size;
    const float* x        = (const float*)d_in[0];
    const float* W_exp    = (const float*)d_in[1];
    const float* b_exp    = (const float*)d_in[2];
    const float* pos_emb  = (const float*)d_in[3];
    const float* ln_ag    = (const float*)d_in[4];
    const float* ln_ab    = (const float*)d_in[5];
    const float* Wqkv     = (const float*)d_in[6];
    const float* Wout     = (const float*)d_in[7];
    const float* ln_fg    = (const float*)d_in[8];
    const float* ln_fb    = (const float*)d_in[9];
    const float* W1       = (const float*)d_in[10];
    const float* b1       = (const float*)d_in[11];
    const float* W2       = (const float*)d_in[12];
    const float* b2       = (const float*)d_in[13];
    const float* lnf_g    = (const float*)d_in[14];
    const float* lnf_b    = (const float*)d_in[15];
    const float* W_logits = (const float*)d_in[16];
    float* out = (float*)d_out;

    // ---- workspace layout ----
    char* ws = (char*)d_ws;
    float*  h    = (float*)ws;                                   // 64 MB
    __bf16* ybuf = (__bf16*)(ws + (size_t)67108864);             // 32 MB
    __bf16* big  = (__bf16*)(ws + (size_t)100663296);            // 192 MB (qkv / ff-hidden)
    __bf16* ao   = (__bf16*)(ws + (size_t)301989888);            // 64 MB
    __bf16* wb   = (__bf16*)(ws + (size_t)369098752);            // 8 MB bf16 transposed weights
    const int nQKV = 4 * kLD * (3 * kINNER);   // 1572864
    const int nOUT = 4 * kINNER * kLD;         // 524288
    const int nW1  = 4 * kLD * kFF;            // 1048576
    const int nW2  = 4 * kFF * kLD;            // 1048576
    __bf16* wbqkv = wb;
    __bf16* wbout = wbqkv + nQKV;
    __bf16* wb1   = wbout + nOUT;
    __bf16* wb2   = wb1 + nW1;

    // ---- convert + transpose weights to bf16 [Nn][K] per layer ----
    transpose_f2bf_kernel<<<(nQKV + 255) / 256, 256, 0, stream>>>(Wqkv, wbqkv, kLD, 3 * kINNER, 4);
    transpose_f2bf_kernel<<<(nOUT + 255) / 256, 256, 0, stream>>>(Wout, wbout, kINNER, kLD, 4);
    transpose_f2bf_kernel<<<(nW1  + 255) / 256, 256, 0, stream>>>(W1,  wb1,  kLD, kFF, 4);
    transpose_f2bf_kernel<<<(nW2  + 255) / 256, 256, 0, stream>>>(W2,  wb2,  kFF, kLD, 4);

    // ---- input expansion ----
    expand_kernel<<<kROWS, 256, 0, stream>>>(x, W_exp, b_exp, pos_emb, h);

    const dim3 blk(256);
    for (int l = 0; l < 4; ++l) {
        // attention sublayer
        ln_kernel<<<kROWS / 8, blk, 0, stream>>>(h, ln_ag + l * kLD, ln_ab + l * kLD, ybuf);
        gemm_bf16_wmma<<<dim3((3 * kINNER) / 128, kROWS / 128), blk, 0, stream>>>(
            ybuf, wbqkv + (size_t)l * kLD * 3 * kINNER, nullptr, nullptr, big,
            kROWS, 3 * kINNER, kLD, 0);
        attn_kernel<<<dim3(kNW, kH, kB), blk, 0, stream>>>(big, ao);
        gemm_bf16_wmma<<<dim3(kLD / 128, kROWS / 128), blk, 0, stream>>>(
            ao, wbout + (size_t)l * kINNER * kLD, nullptr, h, nullptr,
            kROWS, kLD, kINNER, 2);
        // FF sublayer
        ln_kernel<<<kROWS / 8, blk, 0, stream>>>(h, ln_fg + l * kLD, ln_fb + l * kLD, ybuf);
        gemm_bf16_wmma<<<dim3(kFF / 128, kROWS / 128), blk, 0, stream>>>(
            ybuf, wb1 + (size_t)l * kLD * kFF, b1 + (size_t)l * kFF, nullptr, big,
            kROWS, kFF, kLD, 1);
        gemm_bf16_wmma<<<dim3(kLD / 128, kROWS / 128), blk, 0, stream>>>(
            big, wb2 + (size_t)l * kFF * kLD, b2 + (size_t)l * kLD, h, nullptr,
            kROWS, kLD, kFF, 2);
    }

    logits_kernel<<<kROWS / 8, blk, 0, stream>>>(h, lnf_g, lnf_b, W_logits, out);
}
